// AURABlock_51659866636975
// MI455X (gfx1250) — matrix operand order repaired and checked by reading, not compiled, not run on hardware
//
#include <hip/hip_runtime.h>
#include <hip/hip_bf16.h>
#include <math.h>

typedef __bf16 bhalf;
typedef __attribute__((ext_vector_type(16))) __bf16 v16bf;
typedef __attribute__((ext_vector_type(8)))  float  v8f;

#define DIMX   1024
#define ROWSX  2048      // B*T = 2*1024
#define HEADSX 16
#define HDX    64
#define BLKX   128
#define NCHX   8
#define HIDX   4096

// ---------------------------------------------------------------------------
// WMMA helpers (CDNA5 gfx1250, wave32): D = A(16x32 bf16) * B(32x16 bf16) + C(f32)
// ---------------------------------------------------------------------------
static __device__ inline v8f wmma_bf(v16bf a, v16bf b, v8f c) {
  return __builtin_amdgcn_wmma_f32_16x16x32_bf16(false, a, false, b, (short)0, c,
                                                 false, false);
}

static __device__ inline v8f vzero8() {
  v8f z;
#pragma unroll
  for (int i = 0; i < 8; ++i) z[i] = 0.f;
  return z;
}

union frag_u { uint4 u[2]; v16bf v; };

// A-operand fragment (16x32 bf16) from a row-major [M][LD] LDS tile.
// ISA layout: elems 0..7 <-> K = k0+h8..+7 ; elems 8..15 <-> K = k0+16+h8..+7
// -> exactly two contiguous 16B runs in the row. Two ds_load_b128.
template <int LD>
static __device__ inline v16bf frag_a(const bhalf (*S)[LD], int m0, int k0, int lane) {
  const bhalf* p = &S[m0 + (lane & 15)][k0 + ((lane >> 4) << 3)];
  frag_u r;
  r.u[0] = *(const uint4*)(p);
  r.u[1] = *(const uint4*)(p + 16);
  return r.v;
}

// B-operand fragment (32x16 bf16) from an N-major [N][LD] LDS tile:
// lane n takes K = k0 + half*8 .. +7 and k0 + 16 + half*8 .. +7 of row n.
template <int LD>
static __device__ inline v16bf frag_bn(const bhalf (*S)[LD], int n0, int k0, int lane) {
  const bhalf* p = &S[n0 + (lane & 15)][k0 + ((lane >> 4) << 3)];
  frag_u r;
  r.u[0] = *(const uint4*)(p);
  r.u[1] = *(const uint4*)(p + 16);
  return r.v;
}

// B-operand fragment from an N-major fp32 LDS tile (converted on load).
static __device__ inline v16bf frag_bn_f32(const float (*S)[HDX], int n0, int k0,
                                           int lane) {
  const float* p = &S[n0 + (lane & 15)][k0 + ((lane >> 4) << 3)];
  const float4 f0 = *(const float4*)(p);
  const float4 f1 = *(const float4*)(p + 4);
  const float4 f2 = *(const float4*)(p + 16);
  const float4 f3 = *(const float4*)(p + 20);
  v16bf b;
  b[0] = (bhalf)f0.x;  b[1] = (bhalf)f0.y;  b[2] = (bhalf)f0.z;  b[3] = (bhalf)f0.w;
  b[4] = (bhalf)f1.x;  b[5] = (bhalf)f1.y;  b[6] = (bhalf)f1.z;  b[7] = (bhalf)f1.w;
  b[8] = (bhalf)f2.x;  b[9] = (bhalf)f2.y;  b[10] = (bhalf)f2.z; b[11] = (bhalf)f2.w;
  b[12] = (bhalf)f3.x; b[13] = (bhalf)f3.y; b[14] = (bhalf)f3.z; b[15] = (bhalf)f3.w;
  return b;
}

// ---------------------------------------------------------------------------
// Generic bf16 GEMM: C(f32)[M,N] = A[M,K] @ B[K,N], with B supplied TRANSPOSED
// (BT is [N][K] bf16). EPI==2: += residual.
// 256 threads = 8 waves; block tile 128x128; wave tile 32x64; K step 32.
// ---------------------------------------------------------------------------
template <int EPI>
__global__ __launch_bounds__(256) void gemm_bf16_k(
    const bhalf* __restrict__ A, const bhalf* __restrict__ BT,
    const float* __restrict__ resid, float* __restrict__ outF, int M, int N, int K) {
  __shared__ __align__(16) bhalf As[128][32];
  __shared__ __align__(16) bhalf Bs[128][32];   // N-major
  const int tid = threadIdx.x, lane = tid & 31, wave = tid >> 5;
  const int wm = wave & 3, wn = wave >> 2;
  const int bm = blockIdx.y * 128, bn = blockIdx.x * 128;

  v8f acc[2][4];
#pragma unroll
  for (int i = 0; i < 2; ++i)
#pragma unroll
    for (int j = 0; j < 4; ++j) acc[i][j] = vzero8();

  const int r = tid >> 2, c = (tid & 3) << 3;
  for (int kk = 0; kk < K; kk += 32) {
    *(uint4*)(&As[r][c])      = *(const uint4*)(A + (size_t)(bm + r) * K + kk + c);
    *(uint4*)(&As[r + 64][c]) = *(const uint4*)(A + (size_t)(bm + r + 64) * K + kk + c);
    *(uint4*)(&Bs[r][c])      = *(const uint4*)(BT + (size_t)(bn + r) * K + kk + c);
    *(uint4*)(&Bs[r + 64][c]) = *(const uint4*)(BT + (size_t)(bn + r + 64) * K + kk + c);
    if (kk + 32 < K) {  // prefetch next K tile (global_prefetch_b8)
      __builtin_prefetch((const void*)(A + (size_t)(bm + r) * K + kk + 32 + c), 0, 1);
      __builtin_prefetch((const void*)(BT + (size_t)(bn + r) * K + kk + 32 + c), 0, 1);
    }
    __syncthreads();
    v16bf af[2];
#pragma unroll
    for (int i = 0; i < 2; ++i) af[i] = frag_a<32>(As, wm * 32 + i * 16, 0, lane);
#pragma unroll
    for (int j = 0; j < 4; ++j) {
      v16bf bb = frag_bn<32>(Bs, wn * 64 + j * 16, 0, lane);
#pragma unroll
      for (int i = 0; i < 2; ++i) acc[i][j] = wmma_bf(af[i], bb, acc[i][j]);
    }
    __syncthreads();
  }

  const int nloc = lane & 15, h8 = (lane >> 4) << 3;
#pragma unroll
  for (int i = 0; i < 2; ++i)
#pragma unroll
    for (int j = 0; j < 4; ++j)
#pragma unroll
      for (int rr = 0; rr < 8; ++rr) {
        const int row = bm + wm * 32 + i * 16 + h8 + rr;
        const int col = bn + wn * 64 + j * 16 + nloc;
        float v = acc[i][j][rr];
        if (EPI == 2) v += resid[(size_t)row * N + col];
        outF[(size_t)row * N + col] = v;
      }
}

// ---------------------------------------------------------------------------
// Fused SwiGLU up-projection: out(bf16) = silu(A@B1) * (A@B2); B1T/B2T are [N][K]
// ---------------------------------------------------------------------------
__global__ __launch_bounds__(256) void gemm_dual_silu_k(
    const bhalf* __restrict__ A, const bhalf* __restrict__ B1T,
    const bhalf* __restrict__ B2T, bhalf* __restrict__ outB, int M, int N, int K) {
  __shared__ __align__(16) bhalf As[128][32];
  __shared__ __align__(16) bhalf B1s[128][32];
  __shared__ __align__(16) bhalf B2s[128][32];
  const int tid = threadIdx.x, lane = tid & 31, wave = tid >> 5;
  const int wm = wave & 3, wn = wave >> 2;
  const int bm = blockIdx.y * 128, bn = blockIdx.x * 128;

  v8f acc1[2][4], acc2[2][4];
#pragma unroll
  for (int i = 0; i < 2; ++i)
#pragma unroll
    for (int j = 0; j < 4; ++j) { acc1[i][j] = vzero8(); acc2[i][j] = vzero8(); }

  const int r = tid >> 2, c = (tid & 3) << 3;
  for (int kk = 0; kk < K; kk += 32) {
    *(uint4*)(&As[r][c])       = *(const uint4*)(A + (size_t)(bm + r) * K + kk + c);
    *(uint4*)(&As[r + 64][c])  = *(const uint4*)(A + (size_t)(bm + r + 64) * K + kk + c);
    *(uint4*)(&B1s[r][c])      = *(const uint4*)(B1T + (size_t)(bn + r) * K + kk + c);
    *(uint4*)(&B1s[r + 64][c]) = *(const uint4*)(B1T + (size_t)(bn + r + 64) * K + kk + c);
    *(uint4*)(&B2s[r][c])      = *(const uint4*)(B2T + (size_t)(bn + r) * K + kk + c);
    *(uint4*)(&B2s[r + 64][c]) = *(const uint4*)(B2T + (size_t)(bn + r + 64) * K + kk + c);
    __syncthreads();
    v16bf af[2];
#pragma unroll
    for (int i = 0; i < 2; ++i) af[i] = frag_a<32>(As, wm * 32 + i * 16, 0, lane);
#pragma unroll
    for (int j = 0; j < 4; ++j) {
      v16bf b1 = frag_bn<32>(B1s, wn * 64 + j * 16, 0, lane);
      v16bf b2 = frag_bn<32>(B2s, wn * 64 + j * 16, 0, lane);
#pragma unroll
      for (int i = 0; i < 2; ++i) {
        acc1[i][j] = wmma_bf(af[i], b1, acc1[i][j]);
        acc2[i][j] = wmma_bf(af[i], b2, acc2[i][j]);
      }
    }
    __syncthreads();
  }

  const int nloc = lane & 15, h8 = (lane >> 4) << 3;
#pragma unroll
  for (int i = 0; i < 2; ++i)
#pragma unroll
    for (int j = 0; j < 4; ++j)
#pragma unroll
      for (int rr = 0; rr < 8; ++rr) {
        const int row = bm + wm * 32 + i * 16 + h8 + rr;
        const int col = bn + wn * 64 + j * 16 + nloc;
        const float g1 = acc1[i][j][rr], g2 = acc2[i][j][rr];
        const float s = g1 / (1.f + expf(-g1));   // silu
        outB[(size_t)row * N + col] = (bhalf)(s * g2);
      }
}

// ---------------------------------------------------------------------------
// Chunkwise decayed linear attention (RWKV-style), one block per (b,h).
// All matmuls via WMMA, all fragment loads vectorized:
//   rc [l][d] row-major (A-op), kc [l][d] row-major (N-major view for r@k^T),
//   vcT [d->N? no: e][l] == v transposed (N-major B-op for A@v and k^T@v),
//   SstT [e][d] fp32 (N-major B-op for r@S; contiguous C load/store for update).
// LDS = 64KB: rc|kc|vcT (3x16KB) + SstT (16KB); masked-A (32KB) aliases rc+kc.
// ---------------------------------------------------------------------------
__global__ __launch_bounds__(256) void attn_k(const bhalf* __restrict__ R,
                                              const bhalf* __restrict__ Kq,
                                              const bhalf* __restrict__ Vt,
                                              bhalf* __restrict__ Y) {
  __shared__ __align__(16) char smem[64 * 1024];
  bhalf (*rc)[HDX]   = (bhalf(*)[HDX])(smem);
  bhalf (*kc)[HDX]   = (bhalf(*)[HDX])(smem + 16384);
  bhalf (*vcT)[BLKX] = (bhalf(*)[BLKX])(smem + 32768);   // [e=64][l=128]
  float (*SstT)[HDX] = (float(*)[HDX])(smem + 49152);    // [e=64][d=64]
  bhalf (*Am)[BLKX]  = (bhalf(*)[BLKX])(smem);           // aliases rc+kc

  const int bh = blockIdx.x;                     // 0..31
  const int b = bh >> 4, h = bh & 15;
  const int tid = threadIdx.x, lane = tid & 31, wave = tid >> 5;
  const int nloc = lane & 15, h8 = (lane >> 4) << 3;
  const float LOG2D = -0.15200309344504995f;     // log2(0.9)
  const float wL = exp2f(128.0f * LOG2D);        // 0.9^128

  for (int i = tid; i < 64 * 64; i += 256) SstT[i >> 6][i & 63] = 0.f;
  __syncthreads();

  const int m0 = wave * 16;                      // row strip for this wave
  const int mi = wave & 3, nb = (wave >> 2) << 1;// state-update tile assignment

  for (int ch = 0; ch < NCHX; ++ch) {
    const size_t gbase = ((size_t)b * 1024 + (size_t)ch * BLKX) * DIMX + (size_t)h * HDX;
    for (int idx = tid; idx < BLKX * 8; idx += 256) {   // r,k: 128x64 row-major
      const int rr = idx >> 3, cc = (idx & 7) << 3;
      const size_t g = gbase + (size_t)rr * DIMX + cc;
      *(uint4*)(&rc[rr][cc]) = *(const uint4*)(R + g);
      *(uint4*)(&kc[rr][cc]) = *(const uint4*)(Kq + g);
    }
    const size_t gvbase = (size_t)(b * 16 + h) * 64 * 1024 + (size_t)ch * BLKX;
    for (int idx = tid; idx < 64 * 16; idx += 256) {    // vT: 64x128, rows=d
      const int dd = idx >> 4, cc = (idx & 15) << 3;
      *(uint4*)(&vcT[dd][cc]) = *(const uint4*)(Vt + gvbase + (size_t)dd * 1024 + cc);
    }
    __syncthreads();

    // ---- phase 1: inter = (r @ S) row-scaled by w_in;  A = r @ k^T (unmasked);
    //              preload (k * w_out)^T A-fragments for the state update
    v8f accY[4];
#pragma unroll
    for (int j = 0; j < 4; ++j) accY[j] = vzero8();
#pragma unroll
    for (int ks = 0; ks < 2; ++ks) {
      v16bf a = frag_a<HDX>(rc, m0, ks * 32, lane);
#pragma unroll
      for (int j = 0; j < 4; ++j)
        accY[j] = wmma_bf(a, frag_bn_f32(SstT, j * 16, ks * 32, lane), accY[j]);
    }
#pragma unroll
    for (int rr = 0; rr < 8; ++rr) {
      const float wi = exp2f((float)(m0 + h8 + rr + 1) * LOG2D);  // 0.9^(l+1)
#pragma unroll
      for (int j = 0; j < 4; ++j) accY[j][rr] *= wi;
    }

    v8f accA[8];
#pragma unroll
    for (int j = 0; j < 8; ++j) accA[j] = vzero8();
#pragma unroll
    for (int ks = 0; ks < 2; ++ks) {
      v16bf a = frag_a<HDX>(rc, m0, ks * 32, lane);
#pragma unroll
      for (int j = 0; j < 8; ++j)   // B = k^T: N-major view of row-major kc
        accA[j] = wmma_bf(a, frag_bn<HDX>(kc, j * 16, ks * 32, lane), accA[j]);
    }

    v16bf kwf[4];  // A-op for S update: elem (M=d, K=l) = kc[l][d] * 0.9^(127-l)
    {
      const int drow = mi * 16 + nloc;
#pragma unroll
      for (int ks = 0; ks < 4; ++ks) {
        v16bf a;
#pragma unroll
        for (int p = 0; p < 8; ++p) {
          const int l0 = ks * 32 + ((p >> 2) << 4) + h8 + ((p & 3) << 1);
          a[2 * p]     = (bhalf)((float)kc[l0][drow]     * exp2f((float)(127 - l0) * LOG2D));
          a[2 * p + 1] = (bhalf)((float)kc[l0 + 1][drow] * exp2f((float)(126 - l0) * LOG2D));
        }
        kwf[ks] = a;
      }
    }
    __syncthreads();   // rc/kc reads done -> safe to overwrite with masked A

    // masked A -> LDS (bf16), causal decay mask 0.9^(l-m)
#pragma unroll
    for (int j = 0; j < 8; ++j)
#pragma unroll
      for (int rr = 0; rr < 8; ++rr) {
        const int l = m0 + h8 + rr, m = j * 16 + nloc;
        const float v = (l >= m) ? accA[j][rr] * exp2f((float)(l - m) * LOG2D) : 0.f;
        Am[l][m] = (bhalf)v;
      }
    __syncthreads();

    // ---- phase 2: intra = A @ v  (accumulated on top of scaled inter)
#pragma unroll
    for (int ks = 0; ks < 4; ++ks) {
      v16bf a = frag_a<BLKX>(Am, m0, ks * 32, lane);
#pragma unroll
      for (int j = 0; j < 4; ++j)
        accY[j] = wmma_bf(a, frag_bn<BLKX>(vcT, j * 16, ks * 32, lane), accY[j]);
    }

    // ---- state update: S = wL*S + (k*w_out)^T @ v   (2 tiles per wave)
    v8f accS[2];
#pragma unroll
    for (int t = 0; t < 2; ++t) {
      const int ni = nb + t;
      const float* ps = &SstT[ni * 16 + nloc][mi * 16 + h8];  // contiguous over r
      const float4 s0 = *(const float4*)(ps);
      const float4 s1 = *(const float4*)(ps + 4);
      v8f cacc;
      cacc[0] = wL * s0.x; cacc[1] = wL * s0.y; cacc[2] = wL * s0.z; cacc[3] = wL * s0.w;
      cacc[4] = wL * s1.x; cacc[5] = wL * s1.y; cacc[6] = wL * s1.z; cacc[7] = wL * s1.w;
#pragma unroll
      for (int ks = 0; ks < 4; ++ks)
        cacc = wmma_bf(kwf[ks], frag_bn<BLKX>(vcT, ni * 16, ks * 32, lane), cacc);
      accS[t] = cacc;
    }
    __syncthreads();   // all reads of old S done

#pragma unroll
    for (int t = 0; t < 2; ++t) {
      float* ps = &SstT[(nb + t) * 16 + nloc][mi * 16 + h8];
      float4 s0, s1;
      s0.x = accS[t][0]; s0.y = accS[t][1]; s0.z = accS[t][2]; s0.w = accS[t][3];
      s1.x = accS[t][4]; s1.y = accS[t][5]; s1.z = accS[t][6]; s1.w = accS[t][7];
      *(float4*)(ps) = s0;
      *(float4*)(ps + 4) = s1;
    }

    // write y chunk (bf16, [B,T,H*D] layout)
#pragma unroll
    for (int j = 0; j < 4; ++j)
#pragma unroll
      for (int rr = 0; rr < 8; ++rr) {
        const int l = m0 + h8 + rr, e = j * 16 + nloc;
        Y[((size_t)b * 1024 + (size_t)ch * BLKX + l) * DIMX + (size_t)h * HDX + e] =
            (bhalf)accY[j][rr];
      }
    __syncthreads();
  }
}

// ---------------------------------------------------------------------------
// Small kernels: RMSNorm -> bf16, transpose-convert, sigmoid, spike + k-WTA
// ---------------------------------------------------------------------------
__global__ __launch_bounds__(256) void rmsnorm_k(const float* __restrict__ x,
                                                 const float* __restrict__ w,
                                                 bhalf* __restrict__ out) {
  __shared__ float red[256];
  const int row = blockIdx.x;
  const float* xr = x + (size_t)row * DIMX;
  float s = 0.f;
  for (int c = threadIdx.x; c < DIMX; c += 256) { const float v = xr[c]; s += v * v; }
  red[threadIdx.x] = s;
  __syncthreads();
  for (int off = 128; off > 0; off >>= 1) {
    if (threadIdx.x < off) red[threadIdx.x] += red[threadIdx.x + off];
    __syncthreads();
  }
  const float nrm = rsqrtf(red[0] * (1.0f / DIMX) + 1e-5f);
  bhalf* orow = out + (size_t)row * DIMX;
  for (int c = threadIdx.x; c < DIMX; c += 256) orow[c] = (bhalf)(w[c] * xr[c] * nrm);
}

// Tiled transpose + fp32->bf16: in [R][C] fp32 -> out [C][R] bf16.
// blockIdx.z adds a slice offset of R*C elements on both sides.
__global__ void transpose_cvt_k(const float* __restrict__ in, bhalf* __restrict__ out,
                                int R, int C) {
  __shared__ float tile[32][33];
  const int bx = blockIdx.x * 32, by = blockIdx.y * 32;
  const size_t z = (size_t)blockIdx.z * (size_t)R * (size_t)C;
  const int tx = threadIdx.x, ty = threadIdx.y;   // blockDim = (32, 8)
#pragma unroll
  for (int i = 0; i < 32; i += 8)
    tile[ty + i][tx] = in[z + (size_t)(by + ty + i) * C + bx + tx];
  __syncthreads();
#pragma unroll
  for (int i = 0; i < 32; i += 8)
    out[z + (size_t)(bx + ty + i) * R + by + tx] = (bhalf)tile[tx][ty + i];
}

__global__ void sigmoid_bf16_k(const float* __restrict__ in, bhalf* __restrict__ out,
                               int n) {
  const int i = blockIdx.x * 256 + threadIdx.x;
  if (i < n) out[i] = (bhalf)(1.f / (1.f + expf(-in[i])));
}

__global__ void kwta_k(const float* __restrict__ in, bhalf* __restrict__ out) {
  const int unit = blockIdx.x * 256 + threadIdx.x;  // (token, head)
  if (unit >= ROWSX * HEADSX) return;
  const float* p = in + (size_t)unit * HDX;
  float t0 = -1.f, t1 = -1.f, t2 = -1.f, t3 = -1.f;  // thresholded vals are >= 0
  for (int i = 0; i < HDX; ++i) {
    float v = p[i];
    v = (v > 0.5f) ? v : 0.f;
    if (v > t0)      { t3 = t2; t2 = t1; t1 = t0; t0 = v; }
    else if (v > t1) { t3 = t2; t2 = t1; t1 = v; }
    else if (v > t2) { t3 = t2; t2 = v; }
    else if (v > t3) { t3 = v; }
  }
  bhalf* o = out + (size_t)unit * HDX;
  for (int i = 0; i < HDX; ++i) {
    float v = p[i];
    v = (v > 0.5f) ? v : 0.f;
    o[i] = (bhalf)((v >= t3) ? v : 0.f);
  }
}

// ---------------------------------------------------------------------------
// Orchestration
// ---------------------------------------------------------------------------
extern "C" void kernel_launch(void* const* d_in, const int* in_sizes, int n_in,
                              void* d_out, int out_size, void* d_ws, size_t ws_size,
                              hipStream_t stream) {
  const float* x   = (const float*)d_in[0];
  const float* n1w = (const float*)d_in[1];
  const float* Wr  = (const float*)d_in[2];
  const float* Wk  = (const float*)d_in[3];
  const float* Wv  = (const float*)d_in[4];
  const float* Wo  = (const float*)d_in[5];
  const float* n2w = (const float*)d_in[6];
  const float* w1  = (const float*)d_in[7];
  const float* w2  = (const float*)d_in[8];
  const float* w3  = (const float*)d_in[9];
  float* out = (float*)d_out;

  char* ws = (char*)d_ws;
  size_t off = 0;
  auto take = [&](size_t bytes) -> char* {
    char* p = ws + off;
    off += (bytes + 255) & ~(size_t)255;
    return p;
  };

  const size_t ACT = (size_t)ROWSX * DIMX;  // 2M elements
  bhalf* h1  = (bhalf*)take(ACT * 2);
  // All weights stored TRANSPOSED ([N][K]) in bf16 for N-major B staging.
  bhalf* WrT = (bhalf*)take((size_t)DIMX * DIMX * 2);
  bhalf* WkT = (bhalf*)take((size_t)DIMX * DIMX * 2);
  bhalf* WvT = (bhalf*)take((size_t)DIMX * DIMX * 2);
  bhalf* WoT = (bhalf*)take((size_t)DIMX * DIMX * 2);
  bhalf* w1T = (bhalf*)take((size_t)DIMX * HIDX * 2);
  bhalf* w2T = (bhalf*)take((size_t)DIMX * HIDX * 2);
  bhalf* w3T = (bhalf*)take((size_t)HIDX * DIMX * 2);
  // 24MB region, time-multiplexed: {rpre,kpre,vpre} then {hid(16MB), x1(8MB)}
  char* big = take((size_t)24 * 1024 * 1024);
  float* rpre = (float*)big;
  float* kpre = (float*)(big + (size_t)8 * 1024 * 1024);
  float* vpre = (float*)(big + (size_t)16 * 1024 * 1024);
  bhalf* hid  = (bhalf*)big;                                 // after r/k/v consumed
  float* x1   = (float*)(big + (size_t)16 * 1024 * 1024);    // after vpre consumed
  bhalf* rB = (bhalf*)take(ACT * 2);
  bhalf* kB = (bhalf*)take(ACT * 2);
  bhalf* vT = (bhalf*)take(ACT * 2);   // [b,h][d][t] transposed v
  bhalf* yB = (bhalf*)take(ACT * 2);
  bhalf* h2 = (bhalf*)take(ACT * 2);

  // --- weight transpose-converts (fp32 [K][N] -> bf16 [N][K]) ---
  dim3 tb(32, 8);
  transpose_cvt_k<<<dim3(32, 32, 1), tb, 0, stream>>>(Wr, WrT, DIMX, DIMX);
  transpose_cvt_k<<<dim3(32, 32, 1), tb, 0, stream>>>(Wk, WkT, DIMX, DIMX);
  transpose_cvt_k<<<dim3(32, 32, 1), tb, 0, stream>>>(Wv, WvT, DIMX, DIMX);
  transpose_cvt_k<<<dim3(32, 32, 1), tb, 0, stream>>>(Wo, WoT, DIMX, DIMX);
  transpose_cvt_k<<<dim3(HIDX / 32, 32, 1), tb, 0, stream>>>(w1, w1T, DIMX, HIDX);
  transpose_cvt_k<<<dim3(HIDX / 32, 32, 1), tb, 0, stream>>>(w2, w2T, DIMX, HIDX);
  transpose_cvt_k<<<dim3(32, HIDX / 32, 1), tb, 0, stream>>>(w3, w3T, HIDX, DIMX);

  // --- 1. h1 = rmsnorm(x) ---
  rmsnorm_k<<<ROWSX, 256, 0, stream>>>(x, n1w, h1);

  // --- 2. projections r/k/v (WMMA GEMM, fp32 out) ---
  dim3 g_sq(DIMX / 128, ROWSX / 128);  // (8,16)
  gemm_bf16_k<0><<<g_sq, 256, 0, stream>>>(h1, WrT, nullptr, rpre, ROWSX, DIMX, DIMX);
  gemm_bf16_k<0><<<g_sq, 256, 0, stream>>>(h1, WkT, nullptr, kpre, ROWSX, DIMX, DIMX);
  gemm_bf16_k<0><<<g_sq, 256, 0, stream>>>(h1, WvT, nullptr, vpre, ROWSX, DIMX, DIMX);

  // --- 3. activations: sigmoid(r), spike+kWTA(k), v -> transposed bf16 ---
  const int NA = (int)ACT;
  sigmoid_bf16_k<<<(NA + 255) / 256, 256, 0, stream>>>(rpre, rB, NA);
  kwta_k<<<(ROWSX * HEADSX + 255) / 256, 256, 0, stream>>>(kpre, kB);
  // per-batch [t][h*64+d] -> [h*64+d][t]
  transpose_cvt_k<<<dim3(32, 32, 2), tb, 0, stream>>>(vpre, vT, 1024, DIMX);

  // --- 4. chunkwise decayed linear attention (WMMA) ---
  attn_k<<<32, 256, 0, stream>>>(rB, kB, vT, yB);

  // --- 5. x1 = x + y @ Wo ---
  gemm_bf16_k<2><<<g_sq, 256, 0, stream>>>(yB, WoT, x, x1, ROWSX, DIMX, DIMX);

  // --- 6. h2 = rmsnorm(x1) ---
  rmsnorm_k<<<ROWSX, 256, 0, stream>>>(x1, n2w, h2);

  // --- 7. hid = silu(h2@w1) * (h2@w2)  (fused dual WMMA GEMM) ---
  dim3 g_mlp(HIDX / 128, ROWSX / 128);  // (32,16)
  gemm_dual_silu_k<<<g_mlp, 256, 0, stream>>>(h2, w1T, w2T, hid, ROWSX, HIDX, DIMX);

  // --- 8. out = x1 + hid @ w3 ---
  gemm_bf16_k<2><<<g_sq, 256, 0, stream>>>(hid, w3T, x1, out, ROWSX, DIMX, HIDX);
}
